// RecEraser_MF_73289321939066
// MI455X (gfx1250) — matrix operand order repaired and checked by reading, not compiled, not run on hardware
//
#include <hip/hip_runtime.h>
#include <hip/hip_bf16.h>

typedef __attribute__((ext_vector_type(16))) _Float16 v16h;
typedef __attribute__((ext_vector_type(8)))  _Float16 v8h;
typedef __attribute__((ext_vector_type(8)))  float    v8f;
typedef __attribute__((ext_vector_type(4)))  float    v4f;

#define EMBD 64
#define ATTD 32
#define LPART 10

// ---- LDS layout (dynamic shared, ~107 KB of the 320 KB/WGP) ----
// sWt  : f16 [10][64][64]  = trans_W transposed to [l][n][k]      (81920 B)
// sTB  : f32 [10][64]                                             ( 2560 B)
// sWAt : f16 [32][64]      = WA transposed to [a][k]              ( 4096 B)
// sWBt : f16 [32][64]                                             ( 4096 B)
// sBA/sBB/sHA/sHB : f32[32] each                                  (  512 B)
// sE   : f16 [8 waves][16][64] wave-private E staging             (16384 B)
#define OFF_SWT   0
#define OFF_STB   81920
#define OFF_SWAT  84480
#define OFF_SWBT  88576
#define OFF_SBA   92672
#define OFF_SBB   92800
#define OFF_SHA   92928
#define OFF_SHB   93056
#define OFF_SE    93184
#define SMEM_BYTES 109568

// A fragment (16x32 f16) from a global fp32 row, non-temporal (streamed once).
// Lanes 0-15: M=lane, k = kbase+{0..7, 16..23}; lanes 16-31: M=lane-16, k = kbase+{8..15, 24..31}.
__device__ __forceinline__ v16h a_from_global(const float* __restrict__ rowp, int kbase, int hi) {
  const float* p = rowp + kbase + hi * 8;
  v4f x0 = __builtin_nontemporal_load((const v4f*)(p));
  v4f x1 = __builtin_nontemporal_load((const v4f*)(p + 4));
  v4f y0 = __builtin_nontemporal_load((const v4f*)(p + 16));
  v4f y1 = __builtin_nontemporal_load((const v4f*)(p + 20));
  v16h a;
#pragma unroll
  for (int i = 0; i < 4; ++i) {
    a[i]      = (_Float16)x0[i];
    a[4 + i]  = (_Float16)x1[i];
    a[8 + i]  = (_Float16)y0[i];
    a[12 + i] = (_Float16)y1[i];
  }
  return a;
}

// A fragment from LDS f16 row-major [m][k] buffer (rows of EMBD).
__device__ __forceinline__ v16h a_from_lds(const _Float16* __restrict__ eb, int m, int kc, int hi) {
  const _Float16* p = eb + m * EMBD + kc * 32 + hi * 8;
  v8h lo = *(const v8h*)(p);
  v8h hh = *(const v8h*)(p + 16);
  v16h a;
#pragma unroll
  for (int i = 0; i < 8; ++i) { a[i] = lo[i]; a[8 + i] = hh[i]; }
  return a;
}

// B fragment (32x16 f16) from LDS weight stored transposed [n][k], row length EMBD.
// Lanes 0-15: N=n, k = kbase+{0..7, 16..23}; lanes 16-31: k = kbase+{8..15, 24..31}.  Two 8 x f16 loads (16 B each).
__device__ __forceinline__ v16h b_from_lds(const _Float16* __restrict__ wt, int n, int kbase, int hi) {
  const _Float16* p = wt + n * EMBD + kbase + hi * 8;
  v8h lo = *(const v8h*)(p);
  v8h hh = *(const v8h*)(p + 16);
  v16h b;
#pragma unroll
  for (int i = 0; i < 8; ++i) { b[i] = lo[i]; b[8 + i] = hh[i]; }
  return b;
}

__global__ __launch_bounds__(256) void receraser_mf_kernel(
    const int* __restrict__ users, const int* __restrict__ items,
    const float* __restrict__ user_emb, const float* __restrict__ item_emb,
    const float* __restrict__ trans_W, const float* __restrict__ trans_B,
    const float* __restrict__ WA, const float* __restrict__ BA, const float* __restrict__ HA,
    const float* __restrict__ WB, const float* __restrict__ BB, const float* __restrict__ HB,
    float* __restrict__ out, int nB)
{
  extern __shared__ char smem[];
  _Float16* sWt  = (_Float16*)(smem + OFF_SWT);
  float*    sTB  = (float*)   (smem + OFF_STB);
  _Float16* sWAt = (_Float16*)(smem + OFF_SWAT);
  _Float16* sWBt = (_Float16*)(smem + OFF_SWBT);
  float*    sBA  = (float*)   (smem + OFF_SBA);
  float*    sBB  = (float*)   (smem + OFF_SBB);
  float*    sHA  = (float*)   (smem + OFF_SHA);
  float*    sHB  = (float*)   (smem + OFF_SHB);
  _Float16* sE   = (_Float16*)(smem + OFF_SE);

  // ---- stage weights into LDS (coalesced global read, transposed LDS write) ----
  for (int idx = threadIdx.x; idx < LPART * EMBD * EMBD; idx += 256) {
    int l = idx >> 12; int r = idx & 4095; int k = r >> 6; int n = r & 63;
    sWt[(l << 12) + (n << 6) + k] = (_Float16)trans_W[idx];
  }
  for (int idx = threadIdx.x; idx < EMBD * ATTD; idx += 256) {
    int k = idx >> 5; int a = idx & 31;
    sWAt[(a << 6) + k] = (_Float16)WA[idx];
    sWBt[(a << 6) + k] = (_Float16)WB[idx];
  }
  for (int idx = threadIdx.x; idx < LPART * EMBD; idx += 256) sTB[idx] = trans_B[idx];
  if (threadIdx.x < ATTD) {
    sBA[threadIdx.x] = BA[threadIdx.x];
    sBB[threadIdx.x] = BB[threadIdx.x];
    sHA[threadIdx.x] = HA[threadIdx.x];
    sHB[threadIdx.x] = HB[threadIdx.x];
  }
  __syncthreads();

  const int lane = threadIdx.x & 31;
  const int wave = threadIdx.x >> 5;
  const int hi   = lane >> 4;   // half-group select
  const int ln   = lane & 15;   // M (A rows) or N (B/C cols) within tile
  const int rowbase = (blockIdx.x * 8 + wave) * 16;
  if (rowbase >= nB) return;

  int ridx = rowbase + ln; if (ridx >= nB) ridx = nB - 1;
  const float* urow = user_emb + (size_t)users[ridx] * (LPART * EMBD);
  const float* irow = item_emb + (size_t)items[ridx] * (LPART * EMBD);
  _Float16* sEw = sE + wave * (16 * EMBD);

  v8f accU[4], accI[4];
  float ssU[8], ssI[8];
#pragma unroll
  for (int nt = 0; nt < 4; ++nt) {
#pragma unroll
    for (int i = 0; i < 8; ++i) { accU[nt][i] = 0.f; accI[nt][i] = 0.f; }
  }
#pragma unroll
  for (int v = 0; v < 8; ++v) { ssU[v] = 0.f; ssI[v] = 0.f; }

  float ha0u = sHA[ln], ha1u = sHA[16 + ln];
  float ha0i = sHB[ln], ha1i = sHB[16 + ln];

  for (int l = 0; l < LPART; ++l) {
    const _Float16* wl = sWt + (l << 12);
    const float*    tb = sTB + (l << 6);

    // ---- fused transform for BOTH sides: E = e_l * W_l + b_l -----------------
    // One B fragment feeds two WMMAs (user + item chains are independent).
    v8f cu[4], ci[4];
#pragma unroll
    for (int nt = 0; nt < 4; ++nt) {
      float bv = tb[nt * 16 + ln];        // bias depends only on column N
#pragma unroll
      for (int i = 0; i < 8; ++i) { cu[nt][i] = bv; ci[nt][i] = bv; }
    }
#pragma unroll
    for (int kc = 0; kc < 2; ++kc) {
      v16h au = a_from_global(urow, l * EMBD + kc * 32, hi);
      v16h ai = a_from_global(irow, l * EMBD + kc * 32, hi);
#pragma unroll
      for (int nt = 0; nt < 4; ++nt) {
        v16h b = b_from_lds(wl, nt * 16 + ln, kc * 32, hi);
        cu[nt] = __builtin_amdgcn_wmma_f32_16x16x32_f16(
            false, au, false, b, (short)0, cu[nt], false, false);
        ci[nt] = __builtin_amdgcn_wmma_f32_16x16x32_f16(
            false, ai, false, b, (short)0, ci[nt], false, false);
      }
    }

    // ---- per-side attention tail ---------------------------------------------
#pragma unroll
    for (int side = 0; side < 2; ++side) {
      v8f*   c   = side ? ci   : cu;
      v8f*   acc = side ? accI : accU;
      float* ss  = side ? ssI  : ssU;
      const _Float16* wat = side ? sWBt : sWAt;
      const float*    ba  = side ? sBB  : sBA;
      float ha0 = side ? ha0i : ha0u;
      float ha1 = side ? ha1i : ha1u;

      // stage E to wave-private LDS as f16 (C-layout -> row-major [m][k])
#pragma unroll
      for (int nt = 0; nt < 4; ++nt) {
#pragma unroll
        for (int v = 0; v < 8; ++v)
          sEw[(v + hi * 8) * EMBD + nt * 16 + ln] = (_Float16)c[nt][v];
      }

      // T = E*WA + BA via WMMA (16x32 out, K=64)
      v8f t[2];
#pragma unroll
      for (int nt = 0; nt < 2; ++nt) {
        float bv = ba[nt * 16 + ln];
#pragma unroll
        for (int i = 0; i < 8; ++i) t[nt][i] = bv;
      }
#pragma unroll
      for (int kc = 0; kc < 2; ++kc) {
        v16h a = a_from_lds(sEw, ln, kc, hi);
#pragma unroll
        for (int nt = 0; nt < 2; ++nt) {
          v16h b = b_from_lds(wat, nt * 16 + ln, kc * 32, hi);
          t[nt] = __builtin_amdgcn_wmma_f32_16x16x32_f16(
              false, a, false, b, (short)0, t[nt], false, false);
        }
      }

      // s = exp(relu(T) . H); accumulate sum_s and sum_s * E
#pragma unroll
      for (int v = 0; v < 8; ++v) {
        float p = fmaxf(t[0][v], 0.f) * ha0 + fmaxf(t[1][v], 0.f) * ha1;
        p += __shfl_xor(p, 1);
        p += __shfl_xor(p, 2);
        p += __shfl_xor(p, 4);
        p += __shfl_xor(p, 8);   // reduced over the 16-lane half -> uniform
        float s = __expf(p);
        ss[v] += s;
#pragma unroll
        for (int nt = 0; nt < 4; ++nt) acc[nt][v] = fmaf(s, c[nt][v], acc[nt][v]);
      }
    }
  }

  // ---- out[r] = (sum_c accU*accI) / (Su * Si) ----
#pragma unroll
  for (int v = 0; v < 8; ++v) {
    float d = 0.f;
#pragma unroll
    for (int nt = 0; nt < 4; ++nt) d = fmaf(accU[nt][v], accI[nt][v], d);
    d += __shfl_xor(d, 1);
    d += __shfl_xor(d, 2);
    d += __shfl_xor(d, 4);
    d += __shfl_xor(d, 8);
    int orow = rowbase + hi * 8 + v;
    if (ln == 0 && orow < nB) out[orow] = d / (ssU[v] * ssI[v]);
  }
}

extern "C" void kernel_launch(void* const* d_in, const int* in_sizes, int n_in,
                              void* d_out, int out_size, void* d_ws, size_t ws_size,
                              hipStream_t stream) {
  const int*   users    = (const int*)d_in[0];
  const int*   items    = (const int*)d_in[1];
  const float* user_emb = (const float*)d_in[2];
  const float* item_emb = (const float*)d_in[3];
  const float* trans_W  = (const float*)d_in[4];
  const float* trans_B  = (const float*)d_in[5];
  const float* WA       = (const float*)d_in[6];
  const float* BA       = (const float*)d_in[7];
  const float* HA       = (const float*)d_in[8];
  const float* WB       = (const float*)d_in[9];
  const float* BB       = (const float*)d_in[10];
  const float* HB       = (const float*)d_in[11];
  float* out = (float*)d_out;

  int nB = in_sizes[0];
  int blocks = (nB + 127) / 128;   // 8 waves/block * 16 rows/wave

  // Allow >64KB dynamic LDS (gfx1250 supports 320KB/WG). Idempotent, capture-safe.
  (void)hipFuncSetAttribute((const void*)receraser_mf_kernel,
                            hipFuncAttributeMaxDynamicSharedMemorySize, SMEM_BYTES);

  receraser_mf_kernel<<<blocks, 256, SMEM_BYTES, stream>>>(
      users, items, user_emb, item_emb, trans_W, trans_B,
      WA, BA, HA, WB, BB, HB, out, nB);
}
